// BaseMovingLayer_42691974922813
// MI455X (gfx1250) — compile-verified, matrix operand order checked by reading, not run on hardware
//
#include <hip/hip_runtime.h>
#include <hip/hip_bf16.h>

// MEGA MultiHeadEMA (damped EMA conv + SiLU) for MI455X / gfx1250.
//
// The FFT convolution in the reference is mathematically a 16-state linear
// recurrence per channel:  s_n[l] = q_n s_n[l-1] + x[l],  y[l] = sum_n w_n s_n[l]
// with q = 1 - sigmoid(delta)*sigmoid(alpha), w = sigmoid(delta)*sigmoid(alpha)
//                                                  * beta * gamma * sqrt(1/16).
// Chunked-scan formulation (chunk T=16):
//   y_c[i] = sum_j K_d[i][j] x_c[j] + sum_n M_d[i][n] S_c[n]
// where K_d[i][j] = kv_d[i-j] (i>=j), kv_d[t] = sum_n w_n q_n^t,
//       M_d[i][n] = w_n q_n^{i+1}, S_c = state entering chunk c.
// A' = [K_d | M_d] is 16x32, B' = [X_chunk ; S] is 32x16 (cols = 2 chunks x 8
// batches) -> one V_WMMA_F32_16X16X32_F16 per channel per chunk-pair.
//
// Roofline: min traffic 512 MB (x + out) ~ 22 us @ 23.3 TB/s; this pipeline
// moves ~1.0 GB (x twice + 128 MB f16 state array + out) ~ 44 us, with all the
// dense math (4.3 GFLOP) on the matrix pipes instead of 186 TFLOP/s of VALU.

#define L_SEQ 4096
#define B_SZ  8
#define D_CH  2048
#define N_ST  16
#define NCHUNK (L_SEQ / 16)      // 256
#define DTILE 32
#define LTILE 32                 // chunk pair
#define DPAD  33                 // LDS row stride (dwords) -> conflict-free

typedef _Float16 v16h __attribute__((ext_vector_type(16)));
typedef _Float16 v8h  __attribute__((ext_vector_type(8)));
typedef float    v8f  __attribute__((ext_vector_type(8)));

__device__ __forceinline__ float sigf(float v) { return 1.0f / (1.0f + __expf(-v)); }

// ---------------------------------------------------------------------------
// Kernel 0: coefficients. One wave per channel d.
//  - q_ws[d][n] (f32) for the scan kernel.
//  - a_packed: A' = [K_d | M_d] (16x32 f16) pre-swizzled into the WMMA
//    A-operand lane layout (ISA 7.12.2, 16-bit A 16x32):
//      lane<16 : row i=lane, halves e=0..7 -> k=e,    e=8..15 -> k=16+(e-8)
//      lane>=16: row i=lane-16, halves e=0..7 -> k=8+e, e=8..15 -> k=24+(e-8)
// ---------------------------------------------------------------------------
__global__ void __launch_bounds__(256)
ema_coeff_kernel(const float* __restrict__ delta, const float* __restrict__ alpha,
                 const float* __restrict__ beta,  const float* __restrict__ gamma,
                 float* __restrict__ q_ws, _Float16* __restrict__ a_packed) {
    __shared__ float qs[8][16], wsd[8][16], kvs[8][16];
    const int wv = threadIdx.x >> 5;
    const int ln = threadIdx.x & 31;
    const int d  = blockIdx.x * 8 + wv;

    if (ln < 16) {
        const int n = ln;
        const float p = sigf(delta[d * 16 + n]) * sigf(alpha[d * 16 + n]);
        const float q = 1.0f - p;
        const float w = p * beta[d * 16 + n] * gamma[d * 16 + n] * 0.25f; // sqrt(1/16)
        qs[wv][n] = q; wsd[wv][n] = w;
        q_ws[d * 16 + n] = q;
    }
    __syncthreads();
    if (ln < 16) {                 // kernel taps kv[j] = sum_n w_n q_n^j
        const int j = ln;
        float acc = 0.0f;
        #pragma unroll
        for (int n = 0; n < 16; ++n)
            acc += wsd[wv][n] * exp2f((float)j * log2f(qs[wv][n]));
        kvs[wv][j] = acc;
    }
    __syncthreads();

    const int i  = ln & 15;
    const int hi = (ln >= 16) ? 8 : 0;
    v16h av;
    #pragma unroll
    for (int e = 0; e < 8; ++e) {  // K_d block (lower-triangular Toeplitz)
        const int kk = e + hi;
        av[e] = (_Float16)((i >= kk) ? kvs[wv][i - kk] : 0.0f);
    }
    #pragma unroll
    for (int e = 0; e < 8; ++e) {  // M_d block: w_n q_n^{i+1}
        const int n = e + hi;
        av[8 + e] = (_Float16)(wsd[wv][n] * exp2f((float)(i + 1) * log2f(qs[wv][n])));
    }
    *(v16h*)(a_packed + ((size_t)d * 32 + ln) * 16) = av;
}

// ---------------------------------------------------------------------------
// Pass A: register scan, one thread per (b,d). Coalesced x reads (lanes span
// consecutive d). Snapshots the 16-state vector at every chunk boundary into
// S[c][b][d][n] as f16 (exactly the WMMA B-operand rows 16..31).
// ---------------------------------------------------------------------------
__global__ void __launch_bounds__(256)
ema_scan_kernel(const float* __restrict__ x, const float* __restrict__ q_ws,
                _Float16* __restrict__ S) {
    const int t = blockIdx.x * 256 + threadIdx.x;
    const int b = t >> 11;          // 0..7
    const int d = t & (D_CH - 1);   // 0..2047

    float q[16];
    #pragma unroll
    for (int n = 0; n < 16; ++n) q[n] = q_ws[d * 16 + n];
    float s[16];
    #pragma unroll
    for (int n = 0; n < 16; ++n) s[n] = 0.0f;

    // S[0] = 0 (entering chunk 0); d_ws is poisoned, must init.
    {
        v8h z = {};
        _Float16* sp = S + (((size_t)0 * B_SZ + b) * D_CH + d) * 16;
        *(v8h*)sp = z; *(v8h*)(sp + 8) = z;
    }

    for (int c = 0; c < NCHUNK; ++c) {
        const size_t base = ((size_t)(c * 16) * B_SZ + b) * D_CH + d;
        if (c + 1 < NCHUNK)   // gfx1250 global_prefetch_b8 for the next chunk
            __builtin_prefetch(x + base + (size_t)16 * B_SZ * D_CH, 0, 0);
        #pragma unroll
        for (int j = 0; j < 16; ++j) {
            const float xv = x[base + (size_t)j * B_SZ * D_CH];
            #pragma unroll
            for (int n = 0; n < 16; ++n) s[n] = fmaf(q[n], s[n], xv);
        }
        if (c + 1 < NCHUNK) {
            _Float16* sp = S + (((size_t)(c + 1) * B_SZ + b) * D_CH + d) * 16;
            v8h h0, h1;
            #pragma unroll
            for (int n = 0; n < 8; ++n) { h0[n] = (_Float16)s[n]; h1[n] = (_Float16)s[n + 8]; }
            *(v8h*)sp = h0; *(v8h*)(sp + 8) = h1;
        }
    }
}

// ---------------------------------------------------------------------------
// Pass B: block = 32 channels x 32 timesteps (one chunk-pair) x all 8 batches.
// x staged coalesced into LDS; each wave runs one v_wmma_f32_16x16x32_f16 per
// channel (4 channels/wave): D[i,col] = A'(16x32) x B'(32x16),
// cols = (chunk half, batch). Results go back through LDS for a coalesced
// SiLU store. LDS row stride 33 dwords -> bank-conflict free.
// ---------------------------------------------------------------------------
__global__ void __launch_bounds__(256)
ema_wmma_kernel(const float* __restrict__ x, const _Float16* __restrict__ a_packed,
                const _Float16* __restrict__ S, float* __restrict__ out) {
    __shared__ float tile[256 * DPAD];          // [(l_local*8+b)][dl], 33 KB

    const int d0 = blockIdx.x * DTILE;
    const int lp = blockIdx.y;                  // chunk pair index, c0 = 2*lp
    const int tid = threadIdx.x;
    const int wv = tid >> 5, ln = tid & 31;

    // ---- stage x tile (coalesced: lanes span consecutive d) ----
    #pragma unroll
    for (int it = 0; it < 32; ++it) {
        const int idx = it * 256 + tid;
        const int row = idx >> 5;               // (l_local*8 + b), 0..255
        const int dl  = idx & 31;
        tile[row * DPAD + dl] = x[((size_t)lp * 256 + row) * D_CH + d0 + dl];
    }
    __syncthreads();

    // ---- per-lane constants for the B operand / D layout ----
    const int col  = ln & 15;                   // N index
    const int bb   = col & 7;                   // batch of this column
    const int half = col >> 3;                  // 0 -> chunk c0, 1 -> c0+1
    const int hi   = (ln >= 16) ? 8 : 0;        // K sub-block select

    v8f res[4];
    #pragma unroll
    for (int ci = 0; ci < 4; ++ci) {
        const int dl = wv * 4 + ci;
        const int d  = d0 + dl;

        // A operand: pre-swizzled, 32 contiguous bytes per lane.
        const v16h a = *(const v16h*)(a_packed + ((size_t)d * 32 + ln) * 16);

        // B operand: rows 0..15 = x chunk (f32->f16), rows 16..31 = S (f16).
        v16h bvec;
        #pragma unroll
        for (int e = 0; e < 8; ++e) {
            const int j = e + hi;               // time-in-chunk
            bvec[e] = (_Float16)tile[((half * 16 + j) * 8 + bb) * DPAD + dl];
        }
        const int chunk = lp * 2 + half;
        const v8h sv = *(const v8h*)(S + (((size_t)chunk * B_SZ + bb) * D_CH + d) * 16 + hi);
        #pragma unroll
        for (int e = 0; e < 8; ++e) bvec[8 + e] = sv[e];

        v8f cacc = {};
        res[ci] = __builtin_amdgcn_wmma_f32_16x16x32_f16(
            /*neg_a=*/false, a, /*neg_b=*/false, bvec,
            /*c_mod=*/(short)0, cacc, /*reuse_a=*/false, /*reuse_b=*/false);
    }
    __syncthreads();                            // all reads of x tile done

    // ---- scatter D back to LDS (conflict-free: lanes hit distinct rows) ----
    #pragma unroll
    for (int ci = 0; ci < 4; ++ci) {
        const int dl = wv * 4 + ci;
        #pragma unroll
        for (int r = 0; r < 8; ++r) {
            const int i = r + hi;               // D layout: lanes>=16 -> M = r+8
            tile[((half * 16 + i) * 8 + bb) * DPAD + dl] = res[ci][r];
        }
    }
    __syncthreads();

    // ---- SiLU + coalesced store ----
    #pragma unroll
    for (int it = 0; it < 32; ++it) {
        const int idx = it * 256 + tid;
        const int row = idx >> 5;
        const int dl  = idx & 31;
        const float v = tile[row * DPAD + dl];
        out[((size_t)lp * 256 + row) * D_CH + d0 + dl] = v / (1.0f + __expf(-v));
    }
}

// ---------------------------------------------------------------------------
extern "C" void kernel_launch(void* const* d_in, const int* in_sizes, int n_in,
                              void* d_out, int out_size, void* d_ws, size_t ws_size,
                              hipStream_t stream) {
    const float* x     = (const float*)d_in[0];
    const float* delta = (const float*)d_in[1];
    const float* alpha = (const float*)d_in[2];
    const float* beta  = (const float*)d_in[3];
    const float* gamma = (const float*)d_in[4];
    float* out = (float*)d_out;

    // workspace layout (needs ~136.5 MB):
    //   q_ws     : D*16 f32                    = 128 KB
    //   a_packed : D*32 lanes * 16 f16         = 2 MB
    //   S        : 256 * 8 * 2048 * 16 f16     = 128 MB
    char* ws = (char*)d_ws;
    float*    q_ws     = (float*)ws;
    _Float16* a_packed = (_Float16*)(ws + 131072);
    _Float16* S        = (_Float16*)(ws + 2228224);

    ema_coeff_kernel<<<D_CH / 8, 256, 0, stream>>>(delta, alpha, beta, gamma,
                                                   q_ws, a_packed);
    ema_scan_kernel<<<(B_SZ * D_CH) / 256, 256, 0, stream>>>(x, q_ws, S);
    ema_wmma_kernel<<<dim3(D_CH / DTILE, L_SEQ / LTILE), 256, 0, stream>>>(
        x, a_packed, S, out);
}
